// trajectory2seq_31954556682296
// MI455X (gfx1250) — matrix operand order repaired and checked by reading, not compiled, not run on hardware
//
#include <hip/hip_runtime.h>

typedef __bf16 bf16_t;
typedef __attribute__((ext_vector_type(16))) __bf16 v16bf;
typedef __attribute__((ext_vector_type(8)))  float  v8f;

#define BATCH   128
#define TIN     256
#define HDIM    512
#define VOCAB   32
#define MAXLEN  128
#define G3      1536   // 3*H

// A fragment (16x32 bf16): lane<16 holds row m0+lane, K {0..15};
// lane>=16 same rows, K {16..31} (matches B fragment k order).
__device__ __forceinline__ v16bf load_afrag(const bf16_t* __restrict__ p, int k) {
  v16bf a;
#pragma unroll
  for (int i = 0; i < 16; ++i) a[i] = p[k + i];
  return a;
}
// B fragment: lane holds column n0+(lane&15) = a row of W (contiguous 32B);
// K {0..15} (lane<16) / {16..31} (lane>=16).
__device__ __forceinline__ v16bf load_bfrag(const bf16_t* __restrict__ p, int k) {
  v16bf b;
#pragma unroll
  for (int i = 0; i < 16; ++i) b[i] = p[k + i];
  return b;
}
__device__ __forceinline__ v8f wmma_bf16(v16bf a, v16bf b, v8f c) {
  return __builtin_amdgcn_wmma_f32_16x16x32_bf16(false, a, false, b, (short)0, c,
                                                 false, false);
}

// ---------------------------------------------------------------------------
// Generic bf16 GEMM: C[M,N] = A[M,K] * W[N,K]^T + bias[N], fp32 accumulate.
// One wave per 16x16 tile; grid exact multiple of tiles => EXEC all-ones.
// Used for attention query and output logits.
// ---------------------------------------------------------------------------
__global__ void wmma_gemm_bias(const bf16_t* __restrict__ A, int lda,
                               const bf16_t* __restrict__ W, int ldw,
                               const float* __restrict__ bias,
                               float* __restrict__ C, long ldc,
                               int M, int N, int K) {
  const int lane = threadIdx.x & 31;
  const int wid  = (blockIdx.x * blockDim.x + threadIdx.x) >> 5;
  const int tm   = M >> 4;
  const int m0   = (wid % tm) << 4;
  const int n0   = (wid / tm) << 4;

  const bf16_t* __restrict__ ap =
      A + (long)(m0 + (lane & 15)) * lda + ((lane >> 4) << 4);
  const bf16_t* __restrict__ bp =
      W + (long)(n0 + (lane & 15)) * ldw + ((lane >> 4) << 4);

  v8f acc = {0.f, 0.f, 0.f, 0.f, 0.f, 0.f, 0.f, 0.f};
  for (int k = 0; k < K; k += 32)
    acc = wmma_bf16(load_afrag(ap, k), load_bfrag(bp, k), acc);

  const int n  = n0 + (lane & 15);
  const int mb = m0 + ((lane >> 4) << 3);
  const float bv = bias ? bias[n] : 0.f;
#pragma unroll
  for (int v = 0; v < 8; ++v)
    C[(long)(mb + v) * ldc + n] = acc[v] + bv;
}

// ---------------------------------------------------------------------------
// Fully fused GRU cell. Each wave owns one 16x16 tile of hidden columns
// j0..j0+15 and accumulates six WMMA accumulators: (r,z,n) gate columns
// (j, H+j, 2H+j) for both the input path (Ain @ Wih^T) and the hidden path
// (h @ Whh^T). One A-fragment feeds 3 WMMAs. Epilogue does the gate math
// in-register and writes h (fp32 + bf16 [+ optional encoder-output copy]).
//   Ain  == nullptr -> no input GEMM (encoder layer 0; gates come from Gpre)
//   Gpre != nullptr -> precomputed input gates [B, 3H] added in epilogue
// ---------------------------------------------------------------------------
__global__ void gru_cell_fused(const bf16_t* __restrict__ Ain, int Kin,
                               const bf16_t* __restrict__ Wih,   // [3H, Kin]
                               const float* __restrict__ bih,
                               const float* __restrict__ Gpre,   // [B, 3H] or null
                               const bf16_t* __restrict__ Hb,    // h_prev bf16 [B,H]
                               const bf16_t* __restrict__ Whh,   // [3H, H]
                               const float* __restrict__ bhh,
                               const float* __restrict__ hprevf, // h_prev fp32 [B,H]
                               float* __restrict__ hf,
                               bf16_t* __restrict__ hbf,
                               bf16_t* __restrict__ sec, long sec_stride) {
  const int lane = threadIdx.x & 31;
  const int wid  = (blockIdx.x * blockDim.x + threadIdx.x) >> 5;
  const int tm   = BATCH >> 4;                 // 8
  const int m0   = (wid % tm) << 4;
  const int j0   = (wid / tm) << 4;            // hidden-column tile, [0, H)

  const int arow = m0 + (lane & 15);
  const int brow = j0 + (lane & 15);
  const int aoff = (lane >> 4) << 4;
  const int boff = (lane >> 4) << 4;

  const v8f vz = {0.f, 0.f, 0.f, 0.f, 0.f, 0.f, 0.f, 0.f};
  v8f ir = vz, iz = vz, inn = vz, hr = vz, hz = vz, hn = vz;

  {  // hidden path: h @ Whh^T, K = HDIM
    const bf16_t* __restrict__ ap  = Hb + (long)arow * HDIM + aoff;
    const bf16_t* __restrict__ bpr = Whh + (long)brow * HDIM + boff;
    const bf16_t* __restrict__ bpz = Whh + (long)(HDIM + brow) * HDIM + boff;
    const bf16_t* __restrict__ bpn = Whh + (long)(2 * HDIM + brow) * HDIM + boff;
    for (int k = 0; k < HDIM; k += 32) {
      v16bf a = load_afrag(ap, k);
      hr = wmma_bf16(a, load_bfrag(bpr, k), hr);
      hz = wmma_bf16(a, load_bfrag(bpz, k), hz);
      hn = wmma_bf16(a, load_bfrag(bpn, k), hn);
    }
  }
  if (Ain) {  // input path: Ain @ Wih^T, K = Kin (uniform branch)
    const bf16_t* __restrict__ ap  = Ain + (long)arow * Kin + aoff;
    const bf16_t* __restrict__ bpr = Wih + (long)brow * Kin + boff;
    const bf16_t* __restrict__ bpz = Wih + (long)(HDIM + brow) * Kin + boff;
    const bf16_t* __restrict__ bpn = Wih + (long)(2 * HDIM + brow) * Kin + boff;
    for (int k = 0; k < Kin; k += 32) {
      v16bf a = load_afrag(ap, k);
      ir  = wmma_bf16(a, load_bfrag(bpr, k), ir);
      iz  = wmma_bf16(a, load_bfrag(bpz, k), iz);
      inn = wmma_bf16(a, load_bfrag(bpn, k), inn);
    }
  }

  const int j  = j0 + (lane & 15);
  const int mb = m0 + ((lane >> 4) << 3);
  const float bir = bih[j], biz = bih[HDIM + j], bin = bih[2 * HDIM + j];
  const float bhr = bhh[j], bhz = bhh[HDIM + j], bhn = bhh[2 * HDIM + j];
#pragma unroll
  for (int v = 0; v < 8; ++v) {
    const int m = mb + v;
    float gir = ir[v] + bir, giz = iz[v] + biz, gin = inn[v] + bin;
    if (Gpre) {
      const float* gp = Gpre + (long)m * G3;
      gir += gp[j]; giz += gp[HDIM + j]; gin += gp[2 * HDIM + j];
    }
    const float ghr = hr[v] + bhr, ghz = hz[v] + bhz, ghn = hn[v] + bhn;
    const float r = 1.f / (1.f + __expf(-(gir + ghr)));
    const float z = 1.f / (1.f + __expf(-(giz + ghz)));
    const float n = tanhf(gin + r * ghn);
    const float h = (1.f - z) * n + z * hprevf[(long)m * HDIM + j];
    hf[(long)m * HDIM + j]  = h;
    hbf[(long)m * HDIM + j] = (bf16_t)h;
    if (sec) sec[(long)m * sec_stride + j] = (bf16_t)h;
  }
}

// Encoder layer-0 input gates: K=2, pure VALU. Gi[b,n] = x·Wih (no bias; bias
// is added by the fused cell through bih).
__global__ void enc_gi0(const float* __restrict__ x,
                        const float* __restrict__ Wih0,
                        float* __restrict__ Gi, int t) {
  int id = blockIdx.x * blockDim.x + threadIdx.x;   // BATCH*G3
  int b = id / G3, n = id % G3;
  const float* xb = x + ((long)b * TIN + t) * 2;
  Gi[id] = xb[0] * Wih0[n * 2 + 0] + xb[1] * Wih0[n * 2 + 1];
}

// ---------------------------------------------------------------------------
// Attention: scores = q·enc, softmax over T, context, rnn_in = [emb[tok]|ctx].
// One block per batch row; enc is L2-resident bf16.
// ---------------------------------------------------------------------------
__global__ void attn_kernel(const float* __restrict__ query,
                            const bf16_t* __restrict__ enc,
                            const bf16_t* __restrict__ embb,
                            const int* __restrict__ tok,
                            bf16_t* __restrict__ rnn_in,
                            float* __restrict__ attn_out) {
  __shared__ float sw[TIN];
  __shared__ float red[TIN];
  int b = blockIdx.x, t = threadIdx.x;
  const float*  qb = query + (long)b * HDIM;
  const bf16_t* eb = enc + (long)b * TIN * HDIM;

  float s = 0.f;
  for (int h = 0; h < HDIM; ++h) s += qb[h] * (float)eb[t * HDIM + h];

  red[t] = s; __syncthreads();
  for (int o = TIN / 2; o > 0; o >>= 1) {
    if (t < o) red[t] = fmaxf(red[t], red[t + o]);
    __syncthreads();
  }
  float mx = red[0]; __syncthreads();
  float e = __expf(s - mx);
  red[t] = e; __syncthreads();
  for (int o = TIN / 2; o > 0; o >>= 1) {
    if (t < o) red[t] += red[t + o];
    __syncthreads();
  }
  float w = e / red[0];
  sw[t] = w;
  attn_out[(long)b * (MAXLEN * TIN) + t] = w;
  __syncthreads();

  float c0 = 0.f, c1 = 0.f;
  for (int tt = 0; tt < TIN; ++tt) {
    float wv = sw[tt];
    c0 += wv * (float)eb[tt * HDIM + t];
    c1 += wv * (float)eb[tt * HDIM + t + TIN];
  }
  int tk = tok[b];
  bf16_t* rb = rnn_in + (long)b * (2 * HDIM);
  rb[t]              = embb[tk * HDIM + t];
  rb[t + TIN]        = embb[tk * HDIM + t + TIN];
  rb[HDIM + t]       = (bf16_t)c0;
  rb[HDIM + t + TIN] = (bf16_t)c1;
}

// Argmax over V=32 (= one wave), first-max tie-breaking like jnp.argmax.
__global__ void argmax_kernel(const float* __restrict__ logits, int* __restrict__ tok) {
  int b = blockIdx.x, l = threadIdx.x;
  float v = logits[(long)b * (MAXLEN * VOCAB) + l];
  int idx = l;
  for (int o = 16; o > 0; o >>= 1) {
    float ov = __shfl_xor(v, o, 32);
    int   oi = __shfl_xor(idx, o, 32);
    if (ov > v || (ov == v && oi < idx)) { v = ov; idx = oi; }
  }
  if (l == 0) tok[b] = idx;
}

__global__ void cvt_bf16(const float* __restrict__ s, bf16_t* __restrict__ d, int n) {
  int i = blockIdx.x * blockDim.x + threadIdx.x;
  if (i < n) d[i] = (bf16_t)s[i];
}

__global__ void init_state(float* h0f, float* h1f, bf16_t* h0b, bf16_t* h1b,
                           int* tok) {
  int id = blockIdx.x * blockDim.x + threadIdx.x;   // BATCH*HDIM
  h0f[id] = 0.f; h1f[id] = 0.f;
  h0b[id] = (bf16_t)0.f; h1b[id] = (bf16_t)0.f;
  if (id < BATCH) tok[id] = 0;
}

__global__ void copy_hidden(const float* __restrict__ h0f,
                            const float* __restrict__ h1f,
                            float* __restrict__ out) {
  int id = blockIdx.x * blockDim.x + threadIdx.x;   // BATCH*HDIM
  out[id] = h0f[id];
  out[BATCH * HDIM + id] = h1f[id];
}

// ---------------------------------------------------------------------------
extern "C" void kernel_launch(void* const* d_in, const int* in_sizes, int n_in,
                              void* d_out, int out_size, void* d_ws, size_t ws_size,
                              hipStream_t stream) {
  (void)in_sizes; (void)n_in; (void)out_size; (void)ws_size;
  const float* x     = (const float*)d_in[0];
  const float* emb   = (const float*)d_in[1];
  const float* eWih0 = (const float*)d_in[2];
  const float* eWhh0 = (const float*)d_in[3];
  const float* ebih0 = (const float*)d_in[4];
  const float* ebhh0 = (const float*)d_in[5];
  const float* eWih1 = (const float*)d_in[6];
  const float* eWhh1 = (const float*)d_in[7];
  const float* ebih1 = (const float*)d_in[8];
  const float* ebhh1 = (const float*)d_in[9];
  const float* dWih0 = (const float*)d_in[10];
  const float* dWhh0 = (const float*)d_in[11];
  const float* dbih0 = (const float*)d_in[12];
  const float* dbhh0 = (const float*)d_in[13];
  const float* dWih1 = (const float*)d_in[14];
  const float* dWhh1 = (const float*)d_in[15];
  const float* dbih1 = (const float*)d_in[16];
  const float* dbhh1 = (const float*)d_in[17];
  const float* qW    = (const float*)d_in[18];
  const float* qb    = (const float*)d_in[19];
  const float* outW  = (const float*)d_in[20];
  const float* outb  = (const float*)d_in[21];

  float* outf = (float*)d_out;
  float* out_vec = outf;                                   // [B, MAXLEN, V]
  float* out_hid = outf + (long)BATCH * MAXLEN * VOCAB;    // [2, B, H]
  float* out_att = out_hid + 2L * BATCH * HDIM;            // [B, MAXLEN, TIN]

  // ---- workspace carve-out (256B aligned) ----
  char* ws = (char*)d_ws;
  size_t off = 0;
  auto carve = [&](size_t bytes) -> void* {
    void* p = ws + off;
    off = (off + bytes + 255) & ~(size_t)255;
    return p;
  };
  bf16_t* enc_bf = (bf16_t*)carve((size_t)BATCH * TIN * HDIM * 2); // enc outputs
  bf16_t* weWhh0 = (bf16_t*)carve((size_t)G3 * HDIM * 2);
  bf16_t* weWih1 = (bf16_t*)carve((size_t)G3 * HDIM * 2);
  bf16_t* weWhh1 = (bf16_t*)carve((size_t)G3 * HDIM * 2);
  bf16_t* wdWih0 = (bf16_t*)carve((size_t)G3 * 2 * HDIM * 2);
  bf16_t* wdWhh0 = (bf16_t*)carve((size_t)G3 * HDIM * 2);
  bf16_t* wdWih1 = (bf16_t*)carve((size_t)G3 * HDIM * 2);
  bf16_t* wdWhh1 = (bf16_t*)carve((size_t)G3 * HDIM * 2);
  bf16_t* wq     = (bf16_t*)carve((size_t)HDIM * HDIM * 2);
  bf16_t* wout   = (bf16_t*)carve((size_t)VOCAB * HDIM * 2);
  bf16_t* emb_bf = (bf16_t*)carve((size_t)VOCAB * HDIM * 2);
  float*  Gi     = (float*)carve((size_t)BATCH * G3 * 4);  // enc layer-0 input gates
  float*  h0f    = (float*)carve((size_t)BATCH * HDIM * 4);
  float*  h1f    = (float*)carve((size_t)BATCH * HDIM * 4);
  bf16_t* h0b    = (bf16_t*)carve((size_t)BATCH * HDIM * 2);
  bf16_t* h1b    = (bf16_t*)carve((size_t)BATCH * HDIM * 2);
  float*  query  = (float*)carve((size_t)BATCH * HDIM * 4);
  bf16_t* rnn_in = (bf16_t*)carve((size_t)BATCH * 2 * HDIM * 2);
  int*    tok    = (int*)carve((size_t)BATCH * 4);

  auto cvt = [&](const float* s, bf16_t* d, int n) {
    cvt_bf16<<<(n + 255) / 256, 256, 0, stream>>>(s, d, n);
  };
  auto gemm = [&](const bf16_t* A, int lda, const bf16_t* Wm, int ldw,
                  const float* bias, float* C, long ldc, int M, int N, int K) {
    int waves = (M / 16) * (N / 16);
    wmma_gemm_bias<<<waves / 8, 256, 0, stream>>>(A, lda, Wm, ldw, bias, C, ldc,
                                                  M, N, K);
  };
  // fused GRU cell: 8 (M) x 32 (j) tiles = 256 waves = 32 blocks
  auto gru = [&](const bf16_t* Ain, int Kin, const bf16_t* Wih, const float* bih,
                 const float* Gpre, bf16_t* Hb, const bf16_t* Whh,
                 const float* bhh, float* hpf, float* hfo, bf16_t* hbo,
                 bf16_t* sec, long sec_stride) {
    gru_cell_fused<<<32, 256, 0, stream>>>(Ain, Kin, Wih, bih, Gpre, Hb, Whh,
                                           bhh, hpf, hfo, hbo, sec, sec_stride);
  };

  // weight / embedding conversion to bf16
  cvt(eWhh0, weWhh0, G3 * HDIM);
  cvt(eWih1, weWih1, G3 * HDIM);
  cvt(eWhh1, weWhh1, G3 * HDIM);
  cvt(dWih0, wdWih0, G3 * 2 * HDIM);
  cvt(dWhh0, wdWhh0, G3 * HDIM);
  cvt(dWih1, wdWih1, G3 * HDIM);
  cvt(dWhh1, wdWhh1, G3 * HDIM);
  cvt(qW,    wq,     HDIM * HDIM);
  cvt(outW,  wout,   VOCAB * HDIM);
  cvt(emb,   emb_bf, VOCAB * HDIM);

  init_state<<<(BATCH * HDIM) / 256, 256, 0, stream>>>(h0f, h1f, h0b, h1b, tok);

  // ---------------- encoder: 2-layer GRU over T steps ----------------
  for (int t = 0; t < TIN; ++t) {
    enc_gi0<<<(BATCH * G3) / 256, 256, 0, stream>>>(x, eWih0, Gi, t);
    // layer 0: input gates precomputed (K=2), hidden path via WMMA
    gru(nullptr, 0, nullptr, ebih0, Gi, h0b, weWhh0, ebhh0, h0f, h0f, h0b,
        nullptr, 0);
    // layer 1: input = layer-0 hidden; bf16 h1 also lands in enc_bf[:, t, :]
    gru(h0b, HDIM, weWih1, ebih1, nullptr, h1b, weWhh1, ebhh1, h1f, h1f, h1b,
        enc_bf + (long)t * HDIM, (long)TIN * HDIM);
  }

  // ---------------- decoder: MAXLEN autoregressive steps ----------------
  for (int s = 0; s < MAXLEN; ++s) {
    gemm(h1b, HDIM, wq, HDIM, qb, query, HDIM, BATCH, HDIM, HDIM);
    attn_kernel<<<BATCH, TIN, 0, stream>>>(query, enc_bf, emb_bf, tok, rnn_in,
                                           out_att + (long)s * TIN);
    gru(rnn_in, 2 * HDIM, wdWih0, dbih0, nullptr, h0b, wdWhh0, dbhh0, h0f, h0f,
        h0b, nullptr, 0);
    gru(h0b, HDIM, wdWih1, dbih1, nullptr, h1b, wdWhh1, dbhh1, h1f, h1f, h1b,
        nullptr, 0);
    gemm(h1b, HDIM, wout, HDIM, outb, out_vec + (long)s * VOCAB,
         (long)MAXLEN * VOCAB, BATCH, VOCAB, HDIM);
    argmax_kernel<<<BATCH, 32, 0, stream>>>(out_vec + (long)s * VOCAB, tok);
  }

  copy_hidden<<<(BATCH * HDIM) / 256, 256, 0, stream>>>(h0f, h1f, out_hid);
}
